// E_GCL_88476326297737
// MI455X (gfx1250) — matrix operand order repaired and checked by reading, not compiled, not run on hardware
//
#include <hip/hip_runtime.h>
#include <hip/hip_bf16.h>
#include <math.h>

typedef __attribute__((ext_vector_type(16))) _Float16 v16h;
typedef __attribute__((ext_vector_type(8)))  float    v8f;

#define NN 50000
#define NE 800000
#define HH 128
#define DEE 16

// ---------------------------------------------------------------------------
// silu
__device__ __forceinline__ float silu_f(float x) {
    return x / (1.0f + __expf(-x));
}

// ---------------------------------------------------------------------------
// One 16xK @ Kx128 GEMM on a wave: A tile in LDS (row-major, f16, arbitrary
// element stride), B pre-swizzled in global so each lane reads 16 contiguous
// f16 per (kt,nt). Emits 8 column-tiles of the 16x16 f32 D matrix.
// A-fragment k order chosen here (B swizzle matches): lane<16 -> K {0..15}, lane>=16 ->
// K {16..31}; row m = lane&15.
template <int KT, typename F>
__device__ __forceinline__ void gemm16(const _Float16* __restrict__ aLds,
                                       int aStride,
                                       const _Float16* __restrict__ wSwz,
                                       const float* __restrict__ bias,
                                       int lane, F&& emit) {
    const int m  = lane & 15;
    const int kh = lane >> 4;
    v16h a[KT];
#pragma unroll
    for (int kt = 0; kt < KT; ++kt) {
        const _Float16* p = aLds + m * aStride + kt * 32 + kh * 16;
#pragma unroll
        for (int i = 0; i < 8; ++i) {
            a[kt][i]     = p[i];
            a[kt][8 + i] = p[8 + i];
        }
    }
#pragma unroll
    for (int nt = 0; nt < 8; ++nt) {
        v8f acc = {};
#pragma unroll
        for (int kt = 0; kt < KT; ++kt) {
            const v16h* bp =
                (const v16h*)(wSwz + ((((kt * 8 + nt) * 32) + lane) << 4));
            v16h b = *bp;
            acc = __builtin_amdgcn_wmma_f32_16x16x32_f16(
                false, a[kt], false, b, (short)0, acc, false, false);
        }
        float bs = bias[nt * 16 + (lane & 15)];
        emit(nt, acc, bs);
    }
}

// ---------------------------------------------------------------------------
// Weight swizzle: W [K x 128] f32 row-major -> B-fragment order f16.
// out[((kt*8+nt)*32+lane)*16 + i] = W[kt*32 + (lane>>4)*16 + i][nt*16+(lane&15)]
__global__ __launch_bounds__(256) void swizzle_w_kernel(
    const float* __restrict__ W, _Float16* __restrict__ out, int K, int KT) {
    int t = blockIdx.x * blockDim.x + threadIdx.x;
    int total = KT * 8 * 32;
    if (t >= total) return;
    int lane = t & 31;
    int nt   = (t >> 5) & 7;
    int kt   = t >> 8;
    int kh   = lane >> 4;
    int n    = lane & 15;
    _Float16* o = out + t * 16;
#pragma unroll
    for (int i = 0; i < 16; ++i) {
        int k   = kt * 32 + kh * 16 + i;
        float v = (k < K) ? W[k * HH + nt * 16 + n] : 0.0f;
        o[i]    = (_Float16)v;
    }
}

__global__ __launch_bounds__(256) void zero_kernel(float* __restrict__ p, int n) {
    int t = blockIdx.x * blockDim.x + threadIdx.x;
    if (t < n) p[t] = 0.0f;
}

__global__ __launch_bounds__(256) void copy_kernel(const float* __restrict__ s,
                                                   float* __restrict__ d, int n) {
    int t = blockIdx.x * blockDim.x + threadIdx.x;
    if (t < n) d[t] = s[t];
}

// ---------------------------------------------------------------------------
// Fused edge kernel: gather -> mes_mlp (2 layers) -> coord path + scatter
// atomics -> edge_mlp (2 layers). 4 waves/block, 16 edges/wave.
__global__ __launch_bounds__(128) void egcl_edge_kernel(
    const float* __restrict__ h, const float* __restrict__ coord,
    const int* __restrict__ ei, const float* __restrict__ eattr,
    const _Float16* __restrict__ mesW1s, const float* __restrict__ mesB1,
    const _Float16* __restrict__ mesW2s, const float* __restrict__ mesB2,
    const _Float16* __restrict__ coordW1s, const float* __restrict__ coordB1,
    const float* __restrict__ coordW2,
    const _Float16* __restrict__ edgeW1s, const float* __restrict__ edgeB1,
    const _Float16* __restrict__ edgeW2s, const float* __restrict__ edgeB2,
    float* __restrict__ agg, float* __restrict__ coordOut,
    float* __restrict__ edgeOut) {
    const int tid  = threadIdx.x;
    const int lane = tid & 31;
    const int wave = tid >> 5;
    const int e0   = blockIdx.x * 64 + wave * 16;

    __shared__ _Float16 sIn[4][16 * 288];   // mes_in / edge_feat+edge_in
    __shared__ _Float16 sMid[4][16 * 128];  // hidden activations
    __shared__ int      sRC[4][32];         // rows[16], cols[16]
    __shared__ float    sRad[4][16];
    __shared__ float    sCd[4][48];

    _Float16* in_  = sIn[wave];
    _Float16* mid_ = sMid[wave];
    int*   rows = sRC[wave];
    int*   cols = rows + 16;
    float* rad  = sRad[wave];
    float* cd   = sCd[wave];

    // ---- stage (a): indices, radial, coord_diff -----------------------------
    if (lane < 16) {
        int e = e0 + lane;
        int r = ei[e];
        int c = ei[NE + e];
        rows[lane] = r;
        cols[lane] = c;
        float dx = coord[r * 3 + 0] - coord[c * 3 + 0];
        float dy = coord[r * 3 + 1] - coord[c * 3 + 1];
        float dz = coord[r * 3 + 2] - coord[c * 3 + 2];
        float radial = dx * dx + dy * dy + dz * dz;
        float nrm    = sqrtf(radial + 1e-8f) + 1.0f;
        cd[lane * 3 + 0] = dx / nrm;
        cd[lane * 3 + 1] = dy / nrm;
        cd[lane * 3 + 2] = dz / nrm;
        rad[lane]        = radial;
        in_[lane * 288 + 256] = (_Float16)radial;
    }
    __syncthreads();

    // ---- stage (b): gather h[row], h[col] -----------------------------------
    for (int idx = lane; idx < 16 * 256; idx += 32) {
        int m = idx >> 8;
        int c = idx & 255;
        float v = (c < HH) ? h[rows[m] * HH + c] : h[cols[m] * HH + (c - HH)];
        in_[m * 288 + c] = (_Float16)v;
    }
    // ---- stage (c): edge_attr ----------------------------------------------
    for (int idx = lane; idx < 16 * 16; idx += 32) {
        int m = idx >> 4;
        int j = idx & 15;
        in_[m * 288 + 257 + j] = (_Float16)eattr[(e0 + m) * DEE + j];
    }
    // ---- stage (d): zero pad cols 273..287 ---------------------------------
    for (int idx = lane; idx < 240; idx += 32) {
        int m = idx / 15;
        int j = idx - m * 15;
        in_[m * 288 + 273 + j] = (_Float16)0.0f;
    }
    __syncthreads();

    const int kh = lane >> 4;
    const int n  = lane & 15;

    // ---- mes layer 1: t1 = silu(mes_in @ W1 + b1) [K=288] -------------------
    gemm16<9>(in_, 288, mesW1s, mesB1, lane,
              [&](int nt, v8f acc, float bs) {
#pragma unroll
                  for (int r8 = 0; r8 < 8; ++r8) {
                      float x = silu_f(acc[r8] + bs);
                      mid_[(r8 + 8 * kh) * HH + nt * 16 + n] = (_Float16)x;
                  }
              });
    __syncthreads();

    // ---- mes layer 2: ef = silu(t1 @ W2 + b2) -> in_ cols 0..127 ------------
    gemm16<4>(mid_, 128, mesW2s, mesB2, lane,
              [&](int nt, v8f acc, float bs) {
#pragma unroll
                  for (int r8 = 0; r8 < 8; ++r8) {
                      float x = silu_f(acc[r8] + bs);
                      in_[(r8 + 8 * kh) * 288 + nt * 16 + n] = (_Float16)x;
                  }
              });
    // fill edge_in tail: col 128 = radial, 129..144 = edge_attr, 145..159 = 0
    for (int idx = lane; idx < 16 * 32; idx += 32) {
        int m = idx >> 5;
        int j = idx & 31;
        float v;
        if (j == 0)       v = rad[m];
        else if (j <= 16) v = eattr[(e0 + m) * DEE + (j - 1)];
        else              v = 0.0f;
        in_[m * 288 + 128 + j] = (_Float16)v;
    }
    __syncthreads();

    // ---- coord path: c1 = silu(ef @ coordW1 + cb1) --------------------------
    gemm16<4>(in_, 288, coordW1s, coordB1, lane,
              [&](int nt, v8f acc, float bs) {
#pragma unroll
                  for (int r8 = 0; r8 < 8; ++r8) {
                      float x = silu_f(acc[r8] + bs);
                      mid_[(r8 + 8 * kh) * HH + nt * 16 + n] = (_Float16)x;
                  }
              });
    __syncthreads();

    // s = c1 @ coordW2 (H->1); trans = coord_diff * s; scatter to coordOut
    if (lane < 16) {
        float s = 0.0f;
        for (int k2 = 0; k2 < HH; ++k2)
            s += (float)mid_[lane * HH + k2] * coordW2[k2];
        int c = cols[lane];
        atomicAdd(&coordOut[c * 3 + 0], cd[lane * 3 + 0] * s);
        atomicAdd(&coordOut[c * 3 + 1], cd[lane * 3 + 1] * s);
        atomicAdd(&coordOut[c * 3 + 2], cd[lane * 3 + 2] * s);
    }

    // agg[col] += edge_feat (segment_sum via f32 atomics at L2)
    for (int idx = lane; idx < 16 * 128; idx += 32) {
        int m = idx >> 7;
        int f = idx & 127;
        atomicAdd(&agg[cols[m] * HH + f], (float)in_[m * 288 + f]);
    }
    __syncthreads();

    // ---- edge_mlp layer 1: e1 = silu(edge_in @ eW1 + eb1) [K=160] -----------
    gemm16<5>(in_, 288, edgeW1s, edgeB1, lane,
              [&](int nt, v8f acc, float bs) {
#pragma unroll
                  for (int r8 = 0; r8 < 8; ++r8) {
                      float x = silu_f(acc[r8] + bs);
                      mid_[(r8 + 8 * kh) * HH + nt * 16 + n] = (_Float16)x;
                  }
              });
    __syncthreads();

    // ---- edge_mlp layer 2: edge_out = e1 @ eW2 + eb2 -> global --------------
    gemm16<4>(mid_, 128, edgeW2s, edgeB2, lane,
              [&](int nt, v8f acc, float bs) {
#pragma unroll
                  for (int r8 = 0; r8 < 8; ++r8) {
                      int m = r8 + 8 * kh;
                      edgeOut[(size_t)(e0 + m) * HH + nt * 16 + n] =
                          acc[r8] + bs;
                  }
              });
}

// ---------------------------------------------------------------------------
// Node kernel: h_out = h + node_mlp([h, agg])
__global__ __launch_bounds__(128) void egcl_node_kernel(
    const float* __restrict__ h, const float* __restrict__ agg,
    const _Float16* __restrict__ nodeW1s, const float* __restrict__ nodeB1,
    const _Float16* __restrict__ nodeW2s, const float* __restrict__ nodeB2,
    float* __restrict__ hOut) {
    const int tid  = threadIdx.x;
    const int lane = tid & 31;
    const int wave = tid >> 5;
    const int n0   = blockIdx.x * 64 + wave * 16;

    __shared__ _Float16 sIn[4][16 * 256];
    __shared__ _Float16 sMid[4][16 * 128];
    _Float16* in_  = sIn[wave];
    _Float16* mid_ = sMid[wave];

    for (int idx = lane; idx < 16 * 256; idx += 32) {
        int m    = idx >> 8;
        int c    = idx & 255;
        int node = n0 + m;
        float v  = 0.0f;
        if (node < NN)
            v = (c < HH) ? h[node * HH + c] : agg[node * HH + (c - HH)];
        in_[m * 256 + c] = (_Float16)v;
    }
    __syncthreads();

    const int kh = lane >> 4;
    const int n  = lane & 15;

    gemm16<8>(in_, 256, nodeW1s, nodeB1, lane,
              [&](int nt, v8f acc, float bs) {
#pragma unroll
                  for (int r8 = 0; r8 < 8; ++r8) {
                      float x = silu_f(acc[r8] + bs);
                      mid_[(r8 + 8 * kh) * HH + nt * 16 + n] = (_Float16)x;
                  }
              });
    __syncthreads();

    gemm16<4>(mid_, 128, nodeW2s, nodeB2, lane,
              [&](int nt, v8f acc, float bs) {
#pragma unroll
                  for (int r8 = 0; r8 < 8; ++r8) {
                      int node = n0 + r8 + 8 * kh;
                      if (node < NN) {
                          int o   = node * HH + nt * 16 + n;
                          hOut[o] = h[o] + acc[r8] + bs;
                      }
                  }
              });
}

// ---------------------------------------------------------------------------
extern "C" void kernel_launch(void* const* d_in, const int* in_sizes, int n_in,
                              void* d_out, int out_size, void* d_ws,
                              size_t ws_size, hipStream_t stream) {
    const float* h      = (const float*)d_in[0];
    const float* coord  = (const float*)d_in[1];
    const int*   ei     = (const int*)d_in[2];
    const float* eattr  = (const float*)d_in[3];
    const float* mesW1  = (const float*)d_in[4];
    const float* mesB1  = (const float*)d_in[5];
    const float* mesW2  = (const float*)d_in[6];
    const float* mesB2  = (const float*)d_in[7];
    const float* edgeW1 = (const float*)d_in[8];
    const float* edgeB1 = (const float*)d_in[9];
    const float* edgeW2 = (const float*)d_in[10];
    const float* edgeB2 = (const float*)d_in[11];
    const float* nodeW1 = (const float*)d_in[12];
    const float* nodeB1 = (const float*)d_in[13];
    const float* nodeW2 = (const float*)d_in[14];
    const float* nodeB2 = (const float*)d_in[15];
    const float* coordW1 = (const float*)d_in[16];
    const float* coordB1 = (const float*)d_in[17];
    const float* coordW2 = (const float*)d_in[18];

    float* out      = (float*)d_out;
    float* hOut     = out;                       // [N,128]
    float* coordOut = out + (size_t)NN * HH;     // [N,3]
    float* edgeOut  = coordOut + (size_t)NN * 3; // [E,128]

    // workspace layout
    char*  ws   = (char*)d_ws;
    float* agg  = (float*)ws;
    size_t off  = (size_t)NN * HH * sizeof(float);
    _Float16* mesW1s = (_Float16*)(ws + off);  off += 288 * 128 * 2;
    _Float16* mesW2s = (_Float16*)(ws + off);  off += 128 * 128 * 2;
    _Float16* coordW1s = (_Float16*)(ws + off); off += 128 * 128 * 2;
    _Float16* edgeW1s = (_Float16*)(ws + off); off += 160 * 128 * 2;
    _Float16* edgeW2s = (_Float16*)(ws + off); off += 128 * 128 * 2;
    _Float16* nodeW1s = (_Float16*)(ws + off); off += 256 * 128 * 2;
    _Float16* nodeW2s = (_Float16*)(ws + off); off += 128 * 128 * 2;

    auto swz = [&](const float* W, _Float16* o, int K, int KT) {
        int total  = KT * 8 * 32;
        int blocks = (total + 255) / 256;
        swizzle_w_kernel<<<blocks, 256, 0, stream>>>(W, o, K, KT);
    };
    swz(mesW1, mesW1s, 273, 9);
    swz(mesW2, mesW2s, 128, 4);
    swz(coordW1, coordW1s, 128, 4);
    swz(edgeW1, edgeW1s, 145, 5);
    swz(edgeW2, edgeW2s, 128, 4);
    swz(nodeW1, nodeW1s, 256, 8);
    swz(nodeW2, nodeW2s, 128, 4);

    // zero agg; init coordOut = coord
    {
        int n = NN * HH;
        zero_kernel<<<(n + 255) / 256, 256, 0, stream>>>(agg, n);
        int m = NN * 3;
        copy_kernel<<<(m + 255) / 256, 256, 0, stream>>>(coord, coordOut, m);
    }

    // fused edge pass (E divisible by 64)
    egcl_edge_kernel<<<NE / 64, 128, 0, stream>>>(
        h, coord, ei, eattr, mesW1s, mesB1, mesW2s, mesB2, coordW1s, coordB1,
        coordW2, edgeW1s, edgeB1, edgeW2s, edgeB2, agg, coordOut, edgeOut);

    // node pass
    egcl_node_kernel<<<(NN + 63) / 64, 128, 0, stream>>>(
        h, agg, nodeW1s, nodeB1, nodeW2s, nodeB2, hOut);
}
